// FullAttention_17471926960803
// MI455X (gfx1250) — matrix operand order repaired and checked by reading, not compiled, not run on hardware
//
#include <hip/hip_runtime.h>

// CDNA5 / gfx1250 flash-attention forward (causal), f16 WMMA with f32 accumulate.
// B=2, H=16, L=2048, D=64, fp32 in/out.
//
// Block = 4 wave32 waves; each wave owns 16 q rows of a 64-row q-tile.
// K-loop tiles 64 keys at a time: 8 WMMA for S = QK^T, 8 WMMA for O += P V.
// Softmax runs in base-2 with (1/sqrt(D))*log2(e) folded into the Q->f16 cvt.
// Row reductions use single-instruction ds_swizzle XOR exchanges.

typedef __attribute__((ext_vector_type(16))) _Float16 v16h;
typedef __attribute__((ext_vector_type(8)))  _Float16 v8h;
typedef __attribute__((ext_vector_type(4)))  _Float16 v4h;
typedef __attribute__((ext_vector_type(8)))  float    v8f;

#define BQ 64   // q rows per block (16 per wave)
#define BK 64   // keys per k-tile
#define DH 64   // head dim
#define NW 4    // waves per block (wave32)

// group-of-32 ds_swizzle XOR exchange: and_mask=0x1F, or_mask=0, xor_mask=MASK
template <int MASK>
__device__ __forceinline__ float swz_xor(float x) {
    constexpr int pat = 0x001F | (MASK << 10);
    return __builtin_bit_cast(
        float, __builtin_amdgcn_ds_swizzle(__builtin_bit_cast(int, x), pat));
}
__device__ __forceinline__ float red_max16(float x) {
    x = fmaxf(x, swz_xor<1>(x));
    x = fmaxf(x, swz_xor<2>(x));
    x = fmaxf(x, swz_xor<4>(x));
    x = fmaxf(x, swz_xor<8>(x));
    return x;
}
__device__ __forceinline__ float red_sum16(float x) {
    x += swz_xor<1>(x);
    x += swz_xor<2>(x);
    x += swz_xor<4>(x);
    x += swz_xor<8>(x);
    return x;
}

__global__ __launch_bounds__(128) void fa_fwd_causal(
    const float* __restrict__ Q, const float* __restrict__ K,
    const float* __restrict__ V, float* __restrict__ O, int L)
{
    // LDS: K tile [key][d] f16 (8KB), V tile transposed [d][key] f16 (8KB),
    // per-wave P scratch [m][key] f16 (4 x 2KB).
    __shared__ alignas(64) _Float16 Klds[BK * DH];
    __shared__ alignas(64) _Float16 Vlds[DH * BK];
    __shared__ alignas(64) _Float16 Plds[NW][16 * BK];

    const int tid  = threadIdx.x;
    const int wid  = tid >> 5;
    const int lane = tid & 31;
    const int col  = lane & 15;   // N column (C/B layout) / M row (A layout)
    const int hi   = lane >> 4;   // half-wave select

    const size_t base  = (size_t)blockIdx.y * (size_t)L * DH;
    const int    qbase = blockIdx.x * BQ;
    const int    qrow0 = qbase + wid * 16;

    const float* Qp = Q + base;
    const float* Kp = K + base;
    const float* Vp = V + base;
    float*       Op = O + base;

    // softmax in base-2; fold (1/sqrt(D))*log2(e) into the Q fragments
    const float kScl = 0.125f * 1.44269504088896340736f;

    // ---- Q A-fragments (16x32 f16, contiguous k order), loaded once, pre-scaled ----
    // lane holds row M=col; elems e -> d = 32f + 16*hi + e
    v16h aq[2];
    {
        const float* qr = Qp + (size_t)(qrow0 + col) * DH;
        #pragma unroll
        for (int f = 0; f < 2; ++f) {
            const int d0 = 32 * f + hi * 16;
            #pragma unroll
            for (int e = 0; e < 8; ++e) {
                aq[f][e]     = (_Float16)(qr[d0 + e]      * kScl);
                aq[f][8 + e] = (_Float16)(qr[d0 + 8 + e] * kScl);
            }
        }
    }

    const v8f zeroA = {};
    float mx[8], sm[8];
    v8f   o[4];
    #pragma unroll
    for (int v = 0; v < 8; ++v) { mx[v] = -3.0e38f; sm[v] = 0.0f; }
    #pragma unroll
    for (int c = 0; c < 4; ++c) o[c] = zeroA;

    const int kEnd = qbase + BQ;   // causal bound for this block (exclusive)

    for (int kb = 0; kb < kEnd; kb += BK) {
        // ---- cooperative stage of K (f16) and V (f16, transposed) into LDS ----
        #pragma unroll
        for (int r = 0; r < 8; ++r) {
            const int i   = tid + 128 * r;     // 1024 float4's per tile
            const int key = i >> 4;
            const int dp  = (i & 15) << 2;
            const float4 kq = *(const float4*)(Kp + (size_t)(kb + key) * DH + dp);
            v4h kh; kh[0]=(_Float16)kq.x; kh[1]=(_Float16)kq.y; kh[2]=(_Float16)kq.z; kh[3]=(_Float16)kq.w;
            *(v4h*)(&Klds[key * DH + dp]) = kh;
            const float4 vq = *(const float4*)(Vp + (size_t)(kb + key) * DH + dp);
            Vlds[(dp + 0) * BK + key] = (_Float16)vq.x;
            Vlds[(dp + 1) * BK + key] = (_Float16)vq.y;
            Vlds[(dp + 2) * BK + key] = (_Float16)vq.z;
            Vlds[(dp + 3) * BK + key] = (_Float16)vq.w;
        }
        __syncthreads();

        if (kb + BK < kEnd) {  // prefetch next K/V tile toward L2
            __builtin_prefetch(Kp + (size_t)(kb + BK) * DH + tid * 32, 0, 0);
            __builtin_prefetch(Vp + (size_t)(kb + BK) * DH + tid * 32, 0, 0);
        }

        // ---- S = Q K^T for 16 q rows x 64 keys (four 16x16 C-frags) ----
        v8f s[4];
        #pragma unroll
        for (int t = 0; t < 4; ++t) {
            // B frag: lane holds key column N = 16t+col; elems e -> d = 32*ch + 16*hi + e
            const v16h bk0 = *(const v16h*)(&Klds[(t * 16 + col) * DH + 0  + 16 * hi]);
            const v16h bk1 = *(const v16h*)(&Klds[(t * 16 + col) * DH + 32 + 16 * hi]);
            v8f acc = zeroA;
            acc = __builtin_amdgcn_wmma_f32_16x16x32_f16(false, aq[0], false, bk0,
                                                         (short)0, acc, false, false);
            acc = __builtin_amdgcn_wmma_f32_16x16x32_f16(false, aq[1], false, bk1,
                                                         (short)0, acc, false, false);
            s[t] = acc;
        }

        // ---- causal mask: only the diagonal-intersecting tile needs it ----
        if (kb + BK - 1 > qrow0) {
            #pragma unroll
            for (int t = 0; t < 4; ++t) {
                const int key = kb + t * 16 + col;
                #pragma unroll
                for (int v = 0; v < 8; ++v) {
                    const int q = qrow0 + v + 8 * hi;
                    s[t][v] = (key <= q) ? s[t][v] : -3.0e38f;
                }
            }
        }

        // ---- online softmax (base-2) across the 16-lane half-groups ----
        float mnew[8], alpha[8];
        #pragma unroll
        for (int v = 0; v < 8; ++v) {
            float m = fmaxf(fmaxf(s[0][v], s[1][v]), fmaxf(s[2][v], s[3][v]));
            m = red_max16(m);
            m = fmaxf(mx[v], m);
            alpha[v] = exp2f(mx[v] - m);
            mnew[v]  = m;
        }
        #pragma unroll
        for (int v = 0; v < 8; ++v) {
            float r = 0.0f;
            #pragma unroll
            for (int t = 0; t < 4; ++t) {
                const float p = exp2f(s[t][v] - mnew[v]);
                s[t][v] = p;
                r += p;
            }
            r = red_sum16(r);
            sm[v] = sm[v] * alpha[v] + r;
            mx[v] = mnew[v];
        }
        #pragma unroll
        for (int c = 0; c < 4; ++c)
            #pragma unroll
            for (int v = 0; v < 8; ++v) o[c][v] *= alpha[v];

        // ---- P: C layout -> A layout via per-wave LDS scratch ----
        _Float16* pl = &Plds[wid][0];
        #pragma unroll
        for (int t = 0; t < 4; ++t)
            #pragma unroll
            for (int v = 0; v < 8; ++v)
                pl[(v + 8 * hi) * BK + t * 16 + col] = (_Float16)s[t][v];
        asm volatile("s_wait_dscnt 0" ::: "memory");   // same-wave LDS RAW

        v16h pa[2];
        #pragma unroll
        for (int f = 0; f < 2; ++f) {
            const v8h plo = *(const v8h*)(&pl[col * BK + 32 * f + 16 * hi]);
            const v8h phi = *(const v8h*)(&pl[col * BK + 32 * f + 8 + 16 * hi]);
            #pragma unroll
            for (int e = 0; e < 8; ++e) { pa[f][e] = plo[e]; pa[f][8 + e] = phi[e]; }
        }

        // ---- O += P V (4 d-chunks of 16 columns, K = 64 keys = 2 WMMAs) ----
        #pragma unroll
        for (int c = 0; c < 4; ++c) {
            // B frag: lane holds d column N = 16c+col; elems e -> key = 32f + 16*hi + e
            const v16h bv0 = *(const v16h*)(&Vlds[(16 * c + col) * BK + 0  + 16 * hi]);
            const v16h bv1 = *(const v16h*)(&Vlds[(16 * c + col) * BK + 32 + 16 * hi]);
            o[c] = __builtin_amdgcn_wmma_f32_16x16x32_f16(false, pa[0], false, bv0,
                                                          (short)0, o[c], false, false);
            o[c] = __builtin_amdgcn_wmma_f32_16x16x32_f16(false, pa[1], false, bv1,
                                                          (short)0, o[c], false, false);
        }
        __syncthreads();
    }

    // ---- epilogue: normalize by softmax sum, store f32 ----
    #pragma unroll
    for (int v = 0; v < 8; ++v) {
        const float inv = 1.0f / sm[v];
        const int   q   = qrow0 + v + 8 * hi;
        #pragma unroll
        for (int c = 0; c < 4; ++c)
            Op[(size_t)q * DH + 16 * c + col] = o[c][v] * inv;
    }
}

extern "C" void kernel_launch(void* const* d_in, const int* in_sizes, int n_in,
                              void* d_out, int out_size, void* d_ws, size_t ws_size,
                              hipStream_t stream) {
    (void)in_sizes; (void)n_in; (void)out_size; (void)d_ws; (void)ws_size;
    const float* Q = (const float*)d_in[0];
    const float* K = (const float*)d_in[1];
    const float* V = (const float*)d_in[2];
    float*       O = (float*)d_out;

    const int B = 2, H = 16, L = 2048;
    dim3 grid(L / BQ, B * H);   // 32 q-tiles x 32 (b,h) pairs
    dim3 block(32 * NW);        // 4 wave32 waves
    fa_fwd_causal<<<grid, block, 0, stream>>>(Q, K, V, O, L);
}
